// GraphColorNet_73100343378530
// MI455X (gfx1250) — matrix operand order repaired and checked by reading, not compiled, not run on hardware
//
#include <hip/hip_runtime.h>
#include <hip/hip_bf16.h>

// ---------------------------------------------------------------------------
// GraphColorNet forward on MI455X (gfx1250), bf16 WMMA + fused edge pipeline.
// Edge message MLP: weights staged in LDS (w2 via Tensor Data Mover with
// hardware row padding), bank-conflict-free padded layouts, atomics for the
// segment-mean scatter.
// ---------------------------------------------------------------------------

typedef __bf16 bf16_t;
typedef __attribute__((ext_vector_type(16))) __bf16 v16bf;
typedef __attribute__((ext_vector_type(8)))  __bf16 v8bf;
typedef __attribute__((ext_vector_type(8)))  float  v8f;
typedef __attribute__((ext_vector_type(4)))  unsigned int v4u;
typedef __attribute__((ext_vector_type(8)))  int v8i;
typedef __attribute__((ext_vector_type(4)))  int v4i;
typedef __attribute__((address_space(3))) bf16_t* lds_bf16_ptr;

#define HDIM 128
#define W1PAD 296   // 288 + 8  -> 148 dwords/row = 4*odd  (conflict-free)
#define W2PAD 136   // 128 + 8  ->  68 dwords/row = 4*odd  (conflict-free)
#define MSGPAD 136

__device__ __forceinline__ v16bf cat8(v8bf lo, v8bf hi) {
  return __builtin_shufflevector(lo, hi, 0,1,2,3,4,5,6,7,8,9,10,11,12,13,14,15);
}
// A-fragment K-pair: elements [p..p+7] and [p+16..p+23]
__device__ __forceinline__ v16bf load_frag16(const bf16_t* p) {
  v8bf lo = *(const v8bf*)p;
  v8bf hi = *(const v8bf*)(p + 16);
  return cat8(lo, hi);
}
// B-fragment K-pair: elements [p..p+7] and [p+16..p+23], as two 16B loads
__device__ __forceinline__ v16bf load_frag16c(const bf16_t* p) {
  v8bf lo = *(const v8bf*)p;
  v8bf hi = *(const v8bf*)(p + 16);
  return cat8(lo, hi);
}
__device__ __forceinline__ void atomic_add_f32(float* p, float v) {
  __hip_atomic_fetch_add(p, v, __ATOMIC_RELAXED, __HIP_MEMORY_SCOPE_AGENT);
}

// ------------------------------ tiny utilities -----------------------------

__global__ void k_zero(float* __restrict__ p, int n) {
  int i = blockIdx.x * blockDim.x + threadIdx.x;
  if (i < n) p[i] = 0.0f;
}

__global__ void k_convert(const float* __restrict__ s, bf16_t* __restrict__ d, int n) {
  int i = blockIdx.x * blockDim.x + threadIdx.x;
  if (i < n) d[i] = (bf16_t)s[i];
}

// m1_W [128,272] -> padded bf16 [128,288] (last 16 K zero)
__global__ void k_convert_pad288(const float* __restrict__ s, bf16_t* __restrict__ d) {
  int i = blockIdx.x * blockDim.x + threadIdx.x;
  if (i < 128 * 288) {
    int r = i / 288, c = i % 288;
    d[i] = (c < 272) ? (bf16_t)s[r * 272 + c] : (bf16_t)0.0f;
  }
}

// ------------------------------ encoder (K=8) ------------------------------

__global__ void k_encoder(const float* __restrict__ x, const float* __restrict__ W,
                          const float* __restrict__ b, float* __restrict__ y, int N) {
  int i = blockIdx.x * blockDim.x + threadIdx.x;
  if (i >= N * HDIM) return;
  int n = i >> 7, f = i & 127;
  float acc = b[f];
  const float* xr = x + (size_t)n * 8;
  const float* wr = W + (size_t)f * 8;
#pragma unroll
  for (int k = 0; k < 8; ++k) acc += xr[k] * wr[k];
  y[i] = acc;
}

// ------------------------------ BatchNorm ----------------------------------

__global__ __launch_bounds__(256) void k_bn_stats(const float* __restrict__ y,
                                                  float* __restrict__ sum,
                                                  float* __restrict__ sumsq, int N) {
  __shared__ float s1[256], s2[256];
  int f = threadIdx.x & 127;
  int half = threadIdx.x >> 7;
  int r0 = blockIdx.x * 128 + half * 64;
  float a = 0.f, b = 0.f;
  for (int r = r0; r < r0 + 64 && r < N; ++r) {
    float v = y[(size_t)r * HDIM + f];
    a += v; b += v * v;
  }
  s1[threadIdx.x] = a; s2[threadIdx.x] = b;
  __syncthreads();
  if (threadIdx.x < 128) {
    atomic_add_f32(&sum[f],   s1[threadIdx.x] + s1[threadIdx.x + 128]);
    atomic_add_f32(&sumsq[f], s2[threadIdx.x] + s2[threadIdx.x + 128]);
  }
}

__global__ void k_bn_finalize(const float* __restrict__ sum, const float* __restrict__ sumsq,
                              const float* __restrict__ g, const float* __restrict__ be,
                              float* __restrict__ scale, float* __restrict__ shift, int N) {
  int f = threadIdx.x;
  if (f >= HDIM) return;
  float m  = sum[f] / (float)N;
  float v  = sumsq[f] / (float)N - m * m;
  float sc = g[f] * rsqrtf(v + 1e-5f);
  scale[f] = sc;
  shift[f] = be[f] - m * sc;
}

__global__ void k_bn_apply(const float* __restrict__ y, const float* __restrict__ scale,
                           const float* __restrict__ shift, const float* __restrict__ res,
                           float* __restrict__ outF, bf16_t* __restrict__ outB, int total) {
  int i = blockIdx.x * blockDim.x + threadIdx.x;
  if (i >= total) return;
  int f = i & 127;
  float v = fmaf(y[i], scale[f], shift[f]);
  v = v > 0.f ? v : 0.f;
  if (res)  v += res[i];
  if (outF) outF[i] = v;
  outB[i] = (bf16_t)v;
}

// ------------------------------ edge MLP (1->16->16) -----------------------

__global__ void k_edge_mlp(const float* __restrict__ ea_in,
                           const float* __restrict__ w1, const float* __restrict__ b1,
                           const float* __restrict__ w2, const float* __restrict__ b2,
                           bf16_t* __restrict__ out, int E) {
  int e = blockIdx.x * blockDim.x + threadIdx.x;
  if (e >= E) return;
  float a = ea_in[e];
  float h[16];
#pragma unroll
  for (int j = 0; j < 16; ++j) { float v = fmaf(w1[j], a, b1[j]); h[j] = v > 0.f ? v : 0.f; }
#pragma unroll
  for (int k = 0; k < 16; ++k) {
    float v = b2[k];
#pragma unroll
    for (int j = 0; j < 16; ++j) v = fmaf(w2[k * 16 + j], h[j], v);
    out[(size_t)e * 16 + k] = (bf16_t)v;
  }
}

__global__ void k_edge_count(const int* __restrict__ dstIdx, float* __restrict__ cnt, int E) {
  int e = blockIdx.x * blockDim.x + threadIdx.x;
  if (e < E) atomic_add_f32(&cnt[dstIdx[e]], 1.0f);
}

__global__ void k_aggr_fin(const float* __restrict__ aggs, const float* __restrict__ cnt,
                           bf16_t* __restrict__ out, int total) {
  int i = blockIdx.x * blockDim.x + threadIdx.x;
  if (i >= total) return;
  float c = cnt[i >> 7];
  c = c < 1.0f ? 1.0f : c;
  out[i] = (bf16_t)(aggs[i] / c);
}

// ------------------------------ generic WMMA GEMM --------------------------
// out[M,Nout] = act( A[M,K] @ W[Nout,K]^T + bias ), A = concat(A0[:,0:K0], A1)

template <bool BF_OUT, bool RELU>
__global__ __launch_bounds__(256) void k_gemm(
    const bf16_t* __restrict__ A0, const bf16_t* __restrict__ A1, int K0, int K,
    const bf16_t* __restrict__ W, const float* __restrict__ bias,
    float* __restrict__ outF, bf16_t* __restrict__ outB, int M, int Nout) {
  int lane = threadIdx.x & 31;
  int wave = threadIdx.x >> 5;
  int tilesN = Nout >> 4;
  int tile = blockIdx.x * 8 + wave;
  int totalTiles = (M >> 4) * tilesN;
  if (tile >= totalTiles) return;
  int tm = tile / tilesN, tn = tile % tilesN;
  int sub = lane & 15, hiHalf = lane >> 4;
  int row = tm * 16 + sub;
  int n   = tn * 16 + sub;
  const bf16_t* wrow = W + (size_t)n * K + hiHalf * 8;
  int K1 = K - K0;
  v8f c = {};
  for (int kb = 0; kb < (K >> 5); ++kb) {
    int ks = kb * 32;
    const bf16_t* arow = (ks < K0) ? (A0 + (size_t)row * K0 + ks)
                                   : (A1 + (size_t)row * K1 + (ks - K0));
    v16bf a = load_frag16(arow + hiHalf * 8);
    v16bf b = load_frag16(wrow + ks);
    c = __builtin_amdgcn_wmma_f32_16x16x32_bf16(false, a, false, b, (short)0, c, false, false);
  }
  float bs = bias[n];
#pragma unroll
  for (int r = 0; r < 8; ++r) {
    float v = c[r] + bs;
    if (RELU) v = v > 0.f ? v : 0.f;
    int m = tm * 16 + r + hiHalf * 8;
    size_t o = (size_t)m * Nout + n;
    if (BF_OUT) outB[o] = (bf16_t)v; else outF[o] = v;
  }
}

// ------------------------ fused message MLP + scatter-add -------------------
// Per block: stage m1 (pad 296) cooperatively and m2 (pad 136) via TDM into
// LDS; per 16-edge tile (1 wave): gather hn[dst]|hn[src]|ea A-fragments,
// m1 -> ReLU -> LDS(bf16) -> m2 -> +bias -> atomic add into aggr_sum[dst].

__global__ __launch_bounds__(256) void k_message(
    const bf16_t* __restrict__ hn, const bf16_t* __restrict__ ea,
    const int* __restrict__ srcIdx, const int* __restrict__ dstIdx,
    const bf16_t* __restrict__ w1p, const float* __restrict__ b1,   // [128,288],[128]
    const bf16_t* __restrict__ w2,  const float* __restrict__ b2,   // [128,128],[128]
    float* __restrict__ aggr, int E) {
  __shared__ __align__(16) bf16_t w1s[128 * W1PAD];
  __shared__ __align__(16) bf16_t w2s[128 * W2PAD];
  __shared__ __align__(16) bf16_t msgS[8][16 * MSGPAD];

  // ---- stage m1 weights: 128 rows x 36 16B-chunks, padded to 296 ----
  for (int cchunk = threadIdx.x; cchunk < 128 * 36; cchunk += 256) {
    int r = cchunk / 36, c8 = cchunk % 36;
    *(int4*)(w1s + r * W1PAD + c8 * 8) = *(const int4*)(w1p + r * 288 + c8 * 8);
  }
  // ---- stage m2 weights, padded rows 128->136 ----
#if __has_builtin(__builtin_amdgcn_tensor_load_to_lds)
  if ((threadIdx.x >> 5) == 0) {
    // Tensor DMA: 2D 128x128 tile of 2B data, LDS pad 16B every 256B row.
    unsigned ldsa = (unsigned)(size_t)(lds_bf16_ptr)w2s;
    size_t ga = (size_t)w2;
    v4u g0;
    g0[0] = 1u;                                   // count=1 valid descriptor
    g0[1] = ldsa;                                 // lds_addr
    g0[2] = (unsigned)(ga & 0xFFFFFFFFu);         // global_addr[31:0]
    g0[3] = (unsigned)((ga >> 32) & 0x1FFFFFFu)   // global_addr[56:32]
            | (2u << 30);                         // type=2 (image)
    v8i g1;
    g1[0] = (1 << 16)      // data_size = 2B
          | (1 << 20)      // pad_enable
          | (5 << 22)      // pad_interval: 256B
          | (3 << 25);     // pad_amount: 4 dwords = 16B
    g1[1] = (int)(128u << 16);   // tensor_dim0[15:0]=128 (bits 63:48)
    g1[2] = (int)(128u << 16);   // tensor_dim0 hi=0; tensor_dim1[15:0]=128
    g1[3] = (int)(128u << 16);   // tensor_dim1 hi=0; tile_dim0=128
    g1[4] = 128;                 // tile_dim1=128; tile_dim2=0
    g1[5] = 128;                 // tensor_dim0_stride lo32 = 128
    g1[6] = 0; g1[7] = 0;        // dim1 stride unused (2D)
    v4i gz4; gz4[0] = gz4[1] = gz4[2] = gz4[3] = 0;
    v8i gz8;
#pragma unroll
    for (int j = 0; j < 8; ++j) gz8[j] = 0;
    // 6-arg form (clang-23 / therock headers): extra int32x8 group before cpol
    __builtin_amdgcn_tensor_load_to_lds(g0, g1, gz4, gz4, gz8, 0);
    __builtin_amdgcn_s_wait_tensorcnt(0);
  }
#else
  for (int cchunk = threadIdx.x; cchunk < 128 * 16; cchunk += 256) {
    int r = cchunk >> 4, c8 = cchunk & 15;
    *(int4*)(w2s + r * W2PAD + c8 * 8) = *(const int4*)(w2 + r * 128 + c8 * 8);
  }
#endif
  __syncthreads();

  int lane = threadIdx.x & 31;
  int wave = threadIdx.x >> 5;
  int tile = blockIdx.x * 8 + wave;
  if (tile * 16 >= E) return;
  int e0 = tile * 16;
  int sub = lane & 15, hiHalf = lane >> 4;

  int d = dstIdx[e0 + sub];
  int s = srcIdx[e0 + sub];
  const bf16_t* prd = hn + (size_t)d * HDIM;
  const bf16_t* prs = hn + (size_t)s * HDIM;

  v16bf afr[9];
#pragma unroll
  for (int kb = 0; kb < 4; ++kb) afr[kb]     = load_frag16(prd + kb * 32 + hiHalf * 8);
#pragma unroll
  for (int kb = 0; kb < 4; ++kb) afr[4 + kb] = load_frag16(prs + kb * 32 + hiHalf * 8);
  {
    v8bf lo = *(const v8bf*)(ea + (size_t)(e0 + sub) * 16 + hiHalf * 8);
    v8bf z;
#pragma unroll
    for (int j = 0; j < 8; ++j) z[j] = (bf16_t)0.0f;
    afr[8] = cat8(lo, z);
  }

  bf16_t* myl = &msgS[wave][0];
  // m1: 8 column tiles, relu -> LDS (bf16, padded rows)
#pragma unroll
  for (int nt = 0; nt < 8; ++nt) {
    int n = nt * 16 + sub;
    const bf16_t* wrow = w1s + (size_t)n * W1PAD + hiHalf * 8;
    v8f c = {};
#pragma unroll
    for (int kb = 0; kb < 9; ++kb) {
      v16bf b = load_frag16c(wrow + kb * 32);
      c = __builtin_amdgcn_wmma_f32_16x16x32_bf16(false, afr[kb], false, b, (short)0, c, false, false);
    }
    float bs = b1[n];
#pragma unroll
    for (int r = 0; r < 8; ++r) {
      float v = c[r] + bs;
      v = v > 0.f ? v : 0.f;
      int m = r + hiHalf * 8;
      myl[m * MSGPAD + n] = (bf16_t)v;
    }
  }

  asm volatile("s_wait_dscnt 0" ::: "memory");   // wave-local LDS store->load

  v16bf a2[4];
#pragma unroll
  for (int kb = 0; kb < 4; ++kb)
    a2[kb] = load_frag16(myl + sub * MSGPAD + kb * 32 + hiHalf * 8);

  int drows[8];
#pragma unroll
  for (int r = 0; r < 8; ++r) drows[r] = dstIdx[e0 + r + hiHalf * 8];

#pragma unroll
  for (int nt = 0; nt < 8; ++nt) {
    int n = nt * 16 + sub;
    const bf16_t* wrow = w2s + (size_t)n * W2PAD + hiHalf * 8;
    v8f c = {};
#pragma unroll
    for (int kb = 0; kb < 4; ++kb) {
      v16bf b = load_frag16c(wrow + kb * 32);
      c = __builtin_amdgcn_wmma_f32_16x16x32_bf16(false, a2[kb], false, b, (short)0, c, false, false);
    }
    float bs = b2[n];
#pragma unroll
    for (int r = 0; r < 8; ++r)
      atomic_add_f32(&aggr[(size_t)drows[r] * HDIM + n], c[r] + bs);
  }
}

// ------------------------------ output head -------------------------------

__global__ void k_head(const float* __restrict__ t, const float* __restrict__ o2W,
                       const float* __restrict__ o2b, float* __restrict__ out, int N) {
  int n = blockIdx.x * blockDim.x + threadIdx.x;
  if (n >= N) return;
  const float* r = t + (size_t)n * 64;
  float lg[7], mx = -1e30f;
#pragma unroll
  for (int c = 0; c < 7; ++c) {
    float v = o2b[c];
    const float* wr = o2W + c * 64;
#pragma unroll
    for (int j = 0; j < 64; ++j) v = fmaf(wr[j], r[j], v);
    lg[c] = v; mx = fmaxf(mx, v);
  }
  float ssum = 0.f;
#pragma unroll
  for (int c = 0; c < 7; ++c) { lg[c] = __expf(lg[c] - mx); ssum += lg[c]; }
  float inv = 1.0f / ssum;
#pragma unroll
  for (int c = 0; c < 7; ++c) out[(size_t)n * 7 + c] = lg[c] * inv;
}

// ------------------------------ host launcher ------------------------------

static inline size_t align256(size_t x) { return (x + 255) & ~(size_t)255; }
static inline int cdiv(int a, int b) { return (a + b - 1) / b; }

extern "C" void kernel_launch(void* const* d_in, const int* in_sizes, int n_in,
                              void* d_out, int out_size, void* d_ws, size_t ws_size,
                              hipStream_t stream) {
  if (n_in < 65) return;
  const int N = in_sizes[0] / 8;       // 50000
  const int E = in_sizes[2];           // 800000
  const int NT = N * HDIM;

  const float* x     = (const float*)d_in[0];
  const int*   ei    = (const int*)d_in[1];
  const int*   srcI  = ei;
  const int*   dstI  = ei + E;
  const float* eattr = (const float*)d_in[2];
  auto in = [&](int i) { return (const float*)d_in[i]; };

  char* w = (char*)d_ws; size_t off = 0;
  auto alloc = [&](size_t bytes) -> char* { char* p = w + off; off = align256(off + bytes); return p; };
  bf16_t *nodeWbf[3], *m1p[3], *m2bf[3], *u1bf[3], *u2bf[3];
  for (int l = 0; l < 3; ++l) {
    nodeWbf[l] = (bf16_t*)alloc(128 * 128 * 2);
    m1p[l]     = (bf16_t*)alloc(128 * 288 * 2);
    m2bf[l]    = (bf16_t*)alloc(128 * 128 * 2);
    u1bf[l]    = (bf16_t*)alloc(128 * 256 * 2);
    u2bf[l]    = (bf16_t*)alloc(128 * 128 * 2);
  }
  bf16_t* o1bf = (bf16_t*)alloc(64 * 128 * 2);
  float*  hF   = (float*)alloc((size_t)NT * 4);
  float*  y    = (float*)alloc((size_t)NT * 4);
  float*  aggS = (float*)alloc((size_t)NT * 4);
  bf16_t* hB   = (bf16_t*)alloc((size_t)NT * 2);
  bf16_t* hnB  = (bf16_t*)alloc((size_t)NT * 2);
  bf16_t* t1B  = (bf16_t*)alloc((size_t)NT * 2);
  bf16_t* agB  = (bf16_t*)alloc((size_t)NT * 2);
  bf16_t* eaB  = (bf16_t*)alloc((size_t)E * 16 * 2);
  float*  cnt  = (float*)alloc((size_t)N * 4);
  float*  stats = (float*)alloc(512 * 4);
  float *sumB = stats, *sqB = stats + 128, *scB = stats + 256, *shB = stats + 384;
  if (off > ws_size) return;

  const int TPB = 256;
  for (int l = 0; l < 3; ++l) {
    int p = 7 + l * 18;
    k_convert<<<cdiv(16384, TPB), TPB, 0, stream>>>(in(p + 0),  nodeWbf[l], 16384);
    k_convert_pad288<<<cdiv(128 * 288, TPB), TPB, 0, stream>>>(in(p + 8), m1p[l]);
    k_convert<<<cdiv(16384, TPB), TPB, 0, stream>>>(in(p + 10), m2bf[l], 16384);
    k_convert<<<cdiv(32768, TPB), TPB, 0, stream>>>(in(p + 12), u1bf[l], 32768);
    k_convert<<<cdiv(16384, TPB), TPB, 0, stream>>>(in(p + 14), u2bf[l], 16384);
  }
  k_convert<<<cdiv(8192, TPB), TPB, 0, stream>>>(in(61), o1bf, 8192);

  // encoder
  k_encoder<<<cdiv(NT, TPB), TPB, 0, stream>>>(x, in(3), in(4), y, N);
  k_zero<<<2, TPB, 0, stream>>>(stats, 512);
  k_bn_stats<<<cdiv(N, 128), TPB, 0, stream>>>(y, sumB, sqB, N);
  k_bn_finalize<<<1, 128, 0, stream>>>(sumB, sqB, in(5), in(6), scB, shB, N);
  k_bn_apply<<<cdiv(NT, TPB), TPB, 0, stream>>>(y, scB, shB, nullptr, hF, hB, NT);

  const int gemmBlocksH = cdiv((N / 16) * 8, 8);
  for (int l = 0; l < 3; ++l) {
    int p = 7 + l * 18;
    k_gemm<false, false><<<gemmBlocksH, TPB, 0, stream>>>(hB, hB, 128, 128, nodeWbf[l],
                                                          in(p + 1), y, nullptr, N, 128);
    k_zero<<<2, TPB, 0, stream>>>(stats, 512);
    k_bn_stats<<<cdiv(N, 128), TPB, 0, stream>>>(y, sumB, sqB, N);
    k_bn_finalize<<<1, 128, 0, stream>>>(sumB, sqB, in(p + 2), in(p + 3), scB, shB, N);
    k_bn_apply<<<cdiv(NT, TPB), TPB, 0, stream>>>(y, scB, shB, nullptr, nullptr, hnB, NT);

    k_edge_mlp<<<cdiv(E, TPB), TPB, 0, stream>>>(eattr, in(p + 4), in(p + 5),
                                                 in(p + 6), in(p + 7), eaB, E);
    k_zero<<<cdiv(NT, TPB), TPB, 0, stream>>>(aggS, NT);
    k_zero<<<cdiv(N, TPB), TPB, 0, stream>>>(cnt, N);
    k_edge_count<<<cdiv(E, TPB), TPB, 0, stream>>>(dstI, cnt, E);
    k_message<<<cdiv(E / 16, 8), TPB, 0, stream>>>(hnB, eaB, srcI, dstI, m1p[l],
                                                   in(p + 9), m2bf[l], in(p + 11), aggS, E);
    k_aggr_fin<<<cdiv(NT, TPB), TPB, 0, stream>>>(aggS, cnt, agB, NT);

    k_gemm<true, true><<<gemmBlocksH, TPB, 0, stream>>>(hnB, agB, 128, 256, u1bf[l],
                                                        in(p + 13), nullptr, t1B, N, 128);
    k_gemm<false, false><<<gemmBlocksH, TPB, 0, stream>>>(t1B, t1B, 128, 128, u2bf[l],
                                                          in(p + 15), y, nullptr, N, 128);
    k_zero<<<2, TPB, 0, stream>>>(stats, 512);
    k_bn_stats<<<cdiv(N, 128), TPB, 0, stream>>>(y, sumB, sqB, N);
    k_bn_finalize<<<1, 128, 0, stream>>>(sumB, sqB, in(p + 16), in(p + 17), scB, shB, N);
    k_bn_apply<<<cdiv(NT, TPB), TPB, 0, stream>>>(y, scB, shB, (l > 0) ? hF : nullptr,
                                                  hF, hB, NT);
  }

  k_gemm<false, true><<<cdiv((N / 16) * 4, 8), TPB, 0, stream>>>(hB, hB, 128, 128, o1bf,
                                                                 in(62), y, nullptr, N, 64);
  k_head<<<cdiv(N, TPB), TPB, 0, stream>>>(y, in(63), in(64), (float*)d_out, N);
  (void)out_size;
}
